// Discriminator_62019327754336
// MI455X (gfx1250) — matrix operand order repaired and checked by reading, not compiled, not run on hardware
//
#include <hip/hip_runtime.h>
#include <hip/hip_bf16.h>

#define HIDN   1024
#define BATCH  512
#define TSTEPS 100
#define IN_DIM 34
#define KPAD   128              // layer-1 input K padded to one full chunk

#define KC      128             // K elements per staged chunk (4 wmma k-steps)
#define RB      272             // LDS row stride bytes (17*16, odd 16B units -> conflict-free)
#define GSTRIDE (16 * RB)       // per-gate LDS bytes
#define BUFB    (4 * GSTRIDE)   // one chunk buffer (4 gates)

typedef __attribute__((ext_vector_type(16))) __bf16 v16bf;
typedef __attribute__((ext_vector_type(8)))  __bf16 v8bf;
typedef __attribute__((ext_vector_type(8)))  float  v8f;

// ---------------- device helpers ----------------

static __device__ __forceinline__ float sigm(float x) {
    return 1.0f / (1.0f + __expf(-x));
}
static __device__ __forceinline__ float tanh_fast(float x) {
    return 1.0f - 2.0f / (__expf(2.0f * x) + 1.0f);
}

// A fragment: 16x32 bf16, rows m=lane&15; lanes 0-15 hold K=k0+[0..7],k0+16+[0..7],
// lanes 16-31 hold K=k0+8+[0..7],k0+24+[0..7].
static __device__ __forceinline__ v16bf load_a_frag(const __bf16* __restrict__ rowp,
                                                    int k0) {
    v8bf lo  = *(const v8bf*)(rowp + k0);
    v8bf hi8 = *(const v8bf*)(rowp + k0 + 16);
    return __builtin_shufflevector(lo, hi8, 0,1,2,3,4,5,6,7,8,9,10,11,12,13,14,15);
}

static __device__ __forceinline__ v16bf load_b_lds(const char* p) {
    v8bf lo  = *(const v8bf*)(p);
    v8bf hi8 = *(const v8bf*)(p + 32);
    return __builtin_shufflevector(lo, hi8, 0,1,2,3,4,5,6,7,8,9,10,11,12,13,14,15);
}

static __device__ __forceinline__ v8f wmma_bf16(v16bf a, v16bf b, v8f c) {
    return __builtin_amdgcn_wmma_f32_16x16x32_bf16(false, a, false, b,
                                                   (short)0, c, false, false);
}

// Async 16B global->LDS copy (per-lane), tracked by ASYNCcnt.
static __device__ __forceinline__ void async_g2l_b128(unsigned lds_addr, const void* gaddr) {
    asm volatile("global_load_async_to_lds_b128 %0, %1, off"
                 :: "v"(lds_addr), "v"(gaddr) : "memory");
}
static __device__ __forceinline__ void wait_async0() {
    asm volatile("s_wait_asynccnt 0x0" ::: "memory");
}

// ---------------- fused LSTM cell kernel ----------------
// gates[b, g*H + j] = Xa[b,:]*Wa[g*H+j,:] + Hb[b,:]*Wh[g*H+j,:] + bias[g*H+j]
// Wave tile: 32 rows (batch) x 16 cols (hidden) x 4 gates; 8 waves share j0 and
// the LDS-staged weight tiles. grid = (HIDN/16, BATCH/256), block = 256.
__global__ __launch_bounds__(256)
void lstm_cell_wmma(const __bf16* __restrict__ Xa, int KA,       // [BATCH][KA], KA % KC == 0
                    const __bf16* __restrict__ Wa,               // [4*HIDN][KA]
                    const __bf16* __restrict__ Hb,               // [BATCH][HIDN]
                    const __bf16* __restrict__ Wh,               // [4*HIDN][HIDN]
                    const float*  __restrict__ bias,             // [4*HIDN]
                    float*        __restrict__ cst,              // [BATCH][HIDN] in/out
                    __bf16*       __restrict__ hout)             // [BATCH][HIDN]
{
    __shared__ __align__(16) char smem[2 * BUFB];

    const int tid  = threadIdx.x;
    const int lane = tid & 31;
    const int wave = tid >> 5;
    const int j0 = blockIdx.x * 16;                   // hidden column tile
    const int m0 = (blockIdx.y * 8 + wave) * 32;      // batch row base

    const int nc_a = KA / KC;
    const int nc   = nc_a + HIDN / KC;

    // Cooperative async stage of one weight chunk: 4 gates x 16 cols x KC k (bf16).
    // 1024 16-byte units -> 4 per thread.
    auto stage = [&](int ci, int buf) {
        const __bf16* wsrc; int wldk; int kloc;
        if (ci < nc_a) { wsrc = Wa; wldk = KA;   kloc = ci * KC; }
        else           { wsrc = Wh; wldk = HIDN; kloc = (ci - nc_a) * KC; }
        char* dst = smem + buf * BUFB;
#pragma unroll
        for (int r = 0; r < 4; ++r) {
            const int u   = tid + 256 * r;        // 0..1023
            const int g   = u >> 8;               // 4 gates
            const int col = (u >> 4) & 15;        // 16 cols
            const int ku  = u & 15;               // 16 x 16B units per row
            const __bf16* gp = wsrc + (size_t)(g * HIDN + j0 + col) * wldk + kloc + ku * 8;
            unsigned laddr = (unsigned)(size_t)(dst + g * GSTRIDE + col * RB + ku * 16);
            async_g2l_b128(laddr, gp);
        }
    };

    v8f acc[4][2];
#pragma unroll
    for (int g = 0; g < 4; ++g)
#pragma unroll
        for (int mi = 0; mi < 2; ++mi)
            acc[g][mi] = (v8f){0.f,0.f,0.f,0.f,0.f,0.f,0.f,0.f};

    stage(0, 0);

    const int col = lane & 15;
    const int kh  = lane >> 4;

    // Per-wave A row base pointers (lane-dependent part folded in once).
    const int am = lane & 15;
    const int ah = (lane >> 4) * 8;                    // +8 element offset for hi half
    const __bf16* aX0 = Xa + (size_t)(m0 + am) * KA + ah;
    const __bf16* aX1 = Xa + (size_t)(m0 + 16 + am) * KA + ah;
    const __bf16* aH0 = Hb + (size_t)(m0 + am) * HIDN + ah;
    const __bf16* aH1 = Hb + (size_t)(m0 + 16 + am) * HIDN + ah;
    // LDS read base for this lane (col row within each gate block).
    const char* lbase = smem + col * RB;

    for (int ci = 0; ci < nc; ++ci) {
        wait_async0();          // this wave's portion of chunk ci landed
        __syncthreads();        // all waves' portions landed; prev compute done
        if (ci + 1 < nc) stage(ci + 1, (ci + 1) & 1);

        const __bf16 *a0p, *a1p;
        int kga;
        if (ci < nc_a) { a0p = aX0; a1p = aX1; kga = ci * KC; }
        else           { a0p = aH0; a1p = aH1; kga = (ci - nc_a) * KC; }

        const char* bb = lbase + (ci & 1) * BUFB;
#pragma unroll
        for (int ks = 0; ks < KC / 32; ++ks) {
            const int coff = ks * 64 + kh * 16;   // 16B unit within row (k = 32ks + 8h)
            v16bf bm0 = load_b_lds(bb + 0 * GSTRIDE + coff);
            v16bf bm1 = load_b_lds(bb + 1 * GSTRIDE + coff);
            v16bf bm2 = load_b_lds(bb + 2 * GSTRIDE + coff);
            v16bf bm3 = load_b_lds(bb + 3 * GSTRIDE + coff);
            v16bf a0 = load_a_frag(a0p, kga + ks * 32);
            v16bf a1 = load_a_frag(a1p, kga + ks * 32);
            acc[0][0] = wmma_bf16(a0, bm0, acc[0][0]);
            acc[0][1] = wmma_bf16(a1, bm0, acc[0][1]);
            acc[1][0] = wmma_bf16(a0, bm1, acc[1][0]);
            acc[1][1] = wmma_bf16(a1, bm1, acc[1][1]);
            acc[2][0] = wmma_bf16(a0, bm2, acc[2][0]);
            acc[2][1] = wmma_bf16(a1, bm2, acc[2][1]);
            acc[3][0] = wmma_bf16(a0, bm3, acc[3][0]);
            acc[3][1] = wmma_bf16(a1, bm3, acc[3][1]);
        }

        // Software-pipeline the chunk body: operand loads for ks+1 are issued
        // before the WMMAs of ks so DS/VMEM latency hides under matrix math.
        // masks: 0x100 = DS read, 0x20 = VMEM read, 0x8 = MFMA/WMMA.
        __builtin_amdgcn_sched_group_barrier(0x100, 8, 0);  // ks0 B loads
        __builtin_amdgcn_sched_group_barrier(0x020, 4, 0);  // ks0 A loads
        __builtin_amdgcn_sched_group_barrier(0x100, 8, 0);  // ks1 B loads
        __builtin_amdgcn_sched_group_barrier(0x020, 4, 0);  // ks1 A loads
        __builtin_amdgcn_sched_group_barrier(0x008, 8, 0);  // ks0 WMMA
        __builtin_amdgcn_sched_group_barrier(0x100, 8, 0);  // ks2 B loads
        __builtin_amdgcn_sched_group_barrier(0x020, 4, 0);  // ks2 A loads
        __builtin_amdgcn_sched_group_barrier(0x008, 8, 0);  // ks1 WMMA
        __builtin_amdgcn_sched_group_barrier(0x100, 8, 0);  // ks3 B loads
        __builtin_amdgcn_sched_group_barrier(0x020, 4, 0);  // ks3 A loads
        __builtin_amdgcn_sched_group_barrier(0x008, 8, 0);  // ks2 WMMA
        __builtin_amdgcn_sched_group_barrier(0x008, 8, 0);  // ks3 WMMA
    }

    // ---- elementwise cell update ----
    // C/D layout: VGPR r, lanes 0-15 -> M = r; lanes 16-31 -> M = 8 + r; N = lane&15.
    const int colg = j0 + col;
    const int roff = kh * 8;
    const float b_i = bias[0 * HIDN + colg];
    const float b_f = bias[1 * HIDN + colg];
    const float b_g = bias[2 * HIDN + colg];
    const float b_o = bias[3 * HIDN + colg];

#pragma unroll
    for (int mi = 0; mi < 2; ++mi) {
#pragma unroll
        for (int r = 0; r < 8; ++r) {
            const int row = m0 + mi * 16 + roff + r;
            const size_t idx = (size_t)row * HIDN + colg;
            float iv = sigm(acc[0][mi][r] + b_i);
            float fv = sigm(acc[1][mi][r] + b_f);
            float gv = tanh_fast(acc[2][mi][r] + b_g);
            float ov = sigm(acc[3][mi][r] + b_o);
            float cn = fv * cst[idx] + iv * gv;
            cst[idx]  = cn;
            hout[idx] = (__bf16)(ov * tanh_fast(cn));
        }
    }
}

// ---------------- prologue / epilogue kernels ----------------

__global__ void k_cast_bf16(const float* __restrict__ src, __bf16* __restrict__ dst, int n) {
    int i = blockIdx.x * blockDim.x + threadIdx.x;
    if (i < n) dst[i] = (__bf16)src[i];
}

__global__ void k_zero32(unsigned int* __restrict__ p, int n) {
    int i = blockIdx.x * blockDim.x + threadIdx.x;
    if (i < n) p[i] = 0u;
}

// W1x[j4][k] = sum_e W_ih1[j4][e] * enc_W[e][k], padded to KPAD cols (bf16).
__global__ void k_fuse_w1x(const float* __restrict__ Wih1, const float* __restrict__ encW,
                           __bf16* __restrict__ out) {
    __shared__ float rowv[HIDN];
    const int j4 = blockIdx.x;
    for (int e = threadIdx.x; e < HIDN; e += blockDim.x)
        rowv[e] = Wih1[(size_t)j4 * HIDN + e];
    __syncthreads();
    const int k = threadIdx.x;   // blockDim.x == KPAD
    float s = 0.f;
    if (k < IN_DIM) {
        for (int e = 0; e < HIDN; ++e) s += rowv[e] * encW[e * IN_DIM + k];
    }
    out[(size_t)j4 * KPAD + k] = (__bf16)s;
}

// b1c[j4] = W_ih1[j4,:].enc_b + b_ih1[j4] + b_hh1[j4]
__global__ void k_bias1(const float* __restrict__ Wih1, const float* __restrict__ encb,
                        const float* __restrict__ bih, const float* __restrict__ bhh,
                        float* __restrict__ out) {
    int j4 = blockIdx.x * blockDim.x + threadIdx.x;
    if (j4 >= 4 * HIDN) return;
    float s = bih[j4] + bhh[j4];
    for (int e = 0; e < HIDN; ++e) s += Wih1[(size_t)j4 * HIDN + e] * encb[e];
    out[j4] = s;
}

__global__ void k_bias2(const float* __restrict__ bih, const float* __restrict__ bhh,
                        float* __restrict__ out) {
    int j4 = blockIdx.x * blockDim.x + threadIdx.x;
    if (j4 < 4 * HIDN) out[j4] = bih[j4] + bhh[j4];
}

// xpad[t][b][k] = (k < 34) ? x[b][t][k] : 0   (bf16, K padded to KPAD)
__global__ void k_pack_x(const float* __restrict__ x, __bf16* __restrict__ xpad, int n) {
    int i = blockIdx.x * blockDim.x + threadIdx.x;
    if (i >= n) return;
    int k = i % KPAD;
    int b = (i / KPAD) % BATCH;
    int t = i / (KPAD * BATCH);
    float v = (k < IN_DIM) ? x[((size_t)b * TSTEPS + t) * IN_DIM + k] : 0.f;
    xpad[i] = (__bf16)v;
}

// out[b] = h2[b,:] . dec_W + dec_b
__global__ void k_decoder(const __bf16* __restrict__ h2, const float* __restrict__ decW,
                          const float* __restrict__ decb, float* __restrict__ out) {
    int b = blockIdx.x * blockDim.x + threadIdx.x;
    if (b >= BATCH) return;
    float s = decb[0];
    for (int j = 0; j < HIDN; ++j) s += (float)h2[(size_t)b * HIDN + j] * decW[j];
    out[b] = s;
}

// ---------------- launch ----------------

extern "C" void kernel_launch(void* const* d_in, const int* in_sizes, int n_in,
                              void* d_out, int out_size, void* d_ws, size_t ws_size,
                              hipStream_t stream) {
    const float* x     = (const float*)d_in[0];
    const float* encW  = (const float*)d_in[1];
    const float* encb  = (const float*)d_in[2];
    const float* Wih1  = (const float*)d_in[3];
    const float* Whh1  = (const float*)d_in[4];
    const float* bih1  = (const float*)d_in[5];
    const float* bhh1  = (const float*)d_in[6];
    const float* Wih2  = (const float*)d_in[7];
    const float* Whh2  = (const float*)d_in[8];
    const float* bih2  = (const float*)d_in[9];
    const float* bhh2  = (const float*)d_in[10];
    const float* decW  = (const float*)d_in[11];
    const float* decb  = (const float*)d_in[12];
    float* out = (float*)d_out;

    // workspace carve-up (256B aligned; ~46 MB total)
    char* base = (char*)d_ws;
    size_t off = 0;
    auto alloc = [&](size_t bytes) -> void* {
        void* p = base + off;
        off = (off + bytes + 255) & ~(size_t)255;
        return p;
    };
    const size_t WBIG = (size_t)4 * HIDN * HIDN;
    __bf16* whh1_bf = (__bf16*)alloc(WBIG * 2);
    __bf16* wih2_bf = (__bf16*)alloc(WBIG * 2);
    __bf16* whh2_bf = (__bf16*)alloc(WBIG * 2);
    __bf16* w1x_bf  = (__bf16*)alloc((size_t)4 * HIDN * KPAD * 2);
    float*  b1c     = (float*) alloc((size_t)4 * HIDN * 4);
    float*  b2c     = (float*) alloc((size_t)4 * HIDN * 4);
    __bf16* xpad    = (__bf16*)alloc((size_t)TSTEPS * BATCH * KPAD * 2);
    // zero-init region: h buffers (bf16) + cell states (f32), contiguous
    char*   zbase   = base + off;
    __bf16* h1b[2]  = { (__bf16*)alloc((size_t)BATCH * HIDN * 2),
                        (__bf16*)alloc((size_t)BATCH * HIDN * 2) };
    __bf16* h2b[2]  = { (__bf16*)alloc((size_t)BATCH * HIDN * 2),
                        (__bf16*)alloc((size_t)BATCH * HIDN * 2) };
    float*  c1      = (float*) alloc((size_t)BATCH * HIDN * 4);
    float*  c2      = (float*) alloc((size_t)BATCH * HIDN * 4);
    const size_t zbytes = (size_t)((base + off) - zbase);

    // ---- prologue ----
    {
        int n = (int)WBIG, blk = 256, g = (n + blk - 1) / blk;
        k_cast_bf16<<<g, blk, 0, stream>>>(Whh1, whh1_bf, n);
        k_cast_bf16<<<g, blk, 0, stream>>>(Wih2, wih2_bf, n);
        k_cast_bf16<<<g, blk, 0, stream>>>(Whh2, whh2_bf, n);
    }
    k_fuse_w1x<<<4 * HIDN, KPAD, 0, stream>>>(Wih1, encW, w1x_bf);
    k_bias1<<<(4 * HIDN) / 256, 256, 0, stream>>>(Wih1, encb, bih1, bhh1, b1c);
    k_bias2<<<(4 * HIDN) / 256, 256, 0, stream>>>(bih2, bhh2, b2c);
    {
        int n = TSTEPS * BATCH * KPAD;
        k_pack_x<<<(n + 255) / 256, 256, 0, stream>>>(x, xpad, n);
    }
    {
        int n = (int)(zbytes / 4);
        k_zero32<<<(n + 255) / 256, 256, 0, stream>>>((unsigned int*)zbase, n);
    }

    // ---- recurrent scan ----
    dim3 grid(HIDN / 16, BATCH / 256);   // 64 x 2 blocks, 8 waves each
    dim3 blk(256);
    int cur1 = 0, cur2 = 0;
    for (int t = 0; t < TSTEPS; ++t) {
        const __bf16* xt = xpad + (size_t)t * BATCH * KPAD;
        // layer 1: input K=KPAD (fused encoder+W_ih1), recurrent K=1024
        lstm_cell_wmma<<<grid, blk, 0, stream>>>(
            xt, KPAD, w1x_bf, h1b[cur1], whh1_bf, b1c, c1, h1b[cur1 ^ 1]);
        // layer 2: input = fresh h1 (K=1024), recurrent K=1024
        lstm_cell_wmma<<<grid, blk, 0, stream>>>(
            h1b[cur1 ^ 1], HIDN, wih2_bf, h2b[cur2], whh2_bf, b2c, c2, h2b[cur2 ^ 1]);
        cur1 ^= 1;
        cur2 ^= 1;
    }

    // ---- decoder ----
    k_decoder<<<(BATCH + 255) / 256, 256, 0, stream>>>(h2b[cur2], decW, decb, out);

    (void)in_sizes; (void)n_in; (void)out_size; (void)ws_size;
}
